// Model_59519656788080
// MI455X (gfx1250) — matrix operand order repaired and checked by reading, not compiled, not run on hardware
//
#include <hip/hip_runtime.h>

typedef __attribute__((ext_vector_type(16))) _Float16 v16h;
typedef __attribute__((ext_vector_type(8)))  float    v8f;

struct C2 { float x, y; };

__device__ __forceinline__ C2 cmul(C2 a, C2 b){ return C2{a.x*b.x - a.y*b.y, a.x*b.y + a.y*b.x}; }

__device__ __forceinline__ void u3g(const float* p, C2 g[4]){
  float th = p[0], ph = p[1], lm = p[2];
  float ct = __cosf(th*0.5f), st = __sinf(th*0.5f);
  g[0] = C2{ct, 0.f};
  g[1] = C2{-__cosf(lm)*st, -__sinf(lm)*st};
  g[2] = C2{ __cosf(ph)*st,  __sinf(ph)*st};
  g[3] = C2{ __cosf(ph+lm)*ct, __sinf(ph+lm)*ct};
}

// apply 2x2 gate g on wire with bitmask m to a 16-amp column; cm = control mask (0 = none)
__device__ __forceinline__ void applyg(C2 col[16], const C2 g[4], int m, int cm){
  for (int j = 0; j < 16; ++j){
    if (((j & m) == 0) && ((j & cm) == cm)){
      C2 a = col[j], b = col[j | m];
      col[j]     = C2{g[0].x*a.x - g[0].y*a.y + g[1].x*b.x - g[1].y*b.y,
                      g[0].x*a.y + g[0].y*a.x + g[1].x*b.y + g[1].y*b.x};
      col[j | m] = C2{g[2].x*a.x - g[2].y*a.y + g[3].x*b.x - g[3].y*b.y,
                      g[2].x*a.y + g[2].y*a.x + g[3].x*b.y + g[3].y*b.x};
    }
  }
}

__device__ __forceinline__ v8f wmma16(v16h a, v16h b, v8f c){
  // D = A(16x32 f16) * B(32x16 f16) + C(16x16 f32)
  return __builtin_amdgcn_wmma_f32_16x16x32_f16(false, a, false, b, (short)0, c, false, false);
}

__global__ __launch_bounds__(256) void qmnist_kernel(
    const float* __restrict__ x,
    const float* __restrict__ mps_u3, const float* __restrict__ mps_cu3,
    const float* __restrict__ qfc_u3, const float* __restrict__ qfc_cu3,
    const float* __restrict__ Wm, const float* __restrict__ bv,
    float* __restrict__ out, int bsz)
{
  // LDS A-operands, each 32 lanes * 16 halves, already in WMMA A-matrix layout:
  // [0:512)=Ar_mps [512:1024)=Ai_mps [1024:1536)=Ar_qfc [1536:2048)=Ai_qfc
  __shared__ _Float16 sA[2048];
  for (int t = threadIdx.x; t < 2048; t += blockDim.x) sA[t] = (_Float16)0.f;
  __syncthreads();

  const int tid = threadIdx.x, lane = tid & 31, wv = tid >> 5;
  const bool hi = lane >= 16;

  // ---- per-block setup: evolve the fixed 16x16 unitaries column-by-column ----
  if (tid < 32){
    const float* u3p  = hi ? qfc_u3  : mps_u3;
    const float* cup  = hi ? qfc_cu3 : mps_cu3;
    const int    nblk = hi ? 4 : 5;
    const int    mycol = lane & 15;
    C2 col[16];
    for (int j = 0; j < 16; ++j) col[j] = C2{(j == mycol) ? 1.f : 0.f, 0.f};
    const int pc[4] = {0,1,2,3}, pt[4] = {1,2,3,0};  // circular CU3 pairs
    for (int blk = 0; blk < nblk; ++blk){
      for (int w = 0; w < 4; ++w){ C2 g[4]; u3g(u3p + (blk*4 + w)*3, g); applyg(col, g, 8 >> w, 0); }
      for (int k = 0; k < 4; ++k){ C2 g[4]; u3g(cup + (blk*4 + k)*3, g); applyg(col, g, 8 >> pt[k], 8 >> pc[k]); }
    }
    // deposit A in this kernel's contiguous K order (B uses the same): element (M=i, K): lane = i + 16*(K>>4), slot = K&15
    _Float16* Ar = hi ? (sA + 1024) : (sA + 0);
    _Float16* Ai = hi ? (sA + 1536) : (sA + 512);
    const int dlh = 0;
    const int s0  = mycol;
    for (int i = 0; i < 16; ++i){
      int dl = i + dlh;
      Ar[dl*16 + s0] = (_Float16)col[i].x;            // K = mycol
      Ai[dl*16 + s0] = (_Float16)col[i].y;
      if (hi){                                         // K = 16 + mycol (complex packing for QFC)
        Ar[(dl+16)*16 + s0] = (_Float16)(-col[i].y);    // A_r = [Ur | -Ui]
        Ai[(dl+16)*16 + s0] = (_Float16)( col[i].x);    // A_i = [Ui |  Ur]
      }
      // mps K=16..31 stays zero (B upper half is zero there too)
    }
  }
  __syncthreads();

  // load the four A operands (uniform per lane across all waves)
  v16h arm, aim, arq, aiq;
  for (int k = 0; k < 16; ++k){
    arm[k] = sA[lane*16 + k];
    aim[k] = sA[512  + lane*16 + k];
    arq[k] = sA[1024 + lane*16 + k];
    aiq[k] = sA[1536 + lane*16 + k];
  }

  // ---- each wave owns 16 batch elements; lane pair (l, l+16) shares element l&15 ----
  long tile = (long)blockIdx.x * (long)(blockDim.x >> 5) + (long)wv;
  long e0 = tile*16 + (long)(lane & 15);
  bool valid = e0 < (long)bsz;
  long e = valid ? e0 : (long)(bsz - 1);

  // 6x6 avg-pool of 24x24 crop -> xp[4][4]; low lanes cols 0-11, high lanes cols 12-23
  float acc[4][2] = {{0,0},{0,0},{0,0},{0,0}};
  {
    const float* xr = x + (size_t)e * 784 + (hi ? 12 : 0);
    for (int row = 0; row < 24; ++row){
      const float4* p4 = reinterpret_cast<const float4*>(xr + row*28);
      float4 f0 = p4[0], f1 = p4[1], f2 = p4[2];
      int i = row / 6;
      acc[i][0] += f0.x + f0.y + f0.z + f0.w + f1.x + f1.y;
      acc[i][1] += f1.z + f1.w + f2.x + f2.y + f2.z + f2.w;
    }
  }
  float xp[4][4];
  for (int i = 0; i < 4; ++i){
    float a0 = acc[i][0] * (1.f/36.f), a1 = acc[i][1] * (1.f/36.f);
    float b0 = __shfl_xor(a0, 16, 32), b1 = __shfl_xor(a1, 16, 32);
    if (!hi){ xp[i][0]=a0; xp[i][1]=a1; xp[i][2]=b0; xp[i][3]=b1; }
    else    { xp[i][0]=b0; xp[i][1]=b1; xp[i][2]=a0; xp[i][3]=a1; }
  }

  // ---- 4 patch circuits: B = [R ; 0], two WMMAs each, Pauli-Z measure ----
  float meas[4][4];  // [patch][wire]
  for (int pi = 0; pi < 4; ++pi){
    int c = (pi >> 1) * 2, r = (pi & 1) * 2;
    v16h B = {};
    if (!hi){
      float d0 = xp[c][r],   d1 = xp[c][r+1];
      float d2 = xp[c+1][r], d3 = xp[c+1][r+1];
      float c0=__cosf(d0*0.5f), s0=__sinf(d0*0.5f);
      float c1=__cosf(d1*0.5f), s1=__sinf(d1*0.5f);
      float c2=__cosf(d2*0.5f), s2=__sinf(d2*0.5f);
      float c3=__cosf(d3*0.5f), s3=__sinf(d3*0.5f);
      float t01[4] = {c0*c1, c0*s1, s0*c1, s0*s1};
      float t23[4] = {c2*c3, c2*s3, s2*c3, s2*s3};
      for (int j = 0; j < 16; ++j) B[j] = (_Float16)(t01[j >> 2] * t23[j & 3]);
    }
    v8f z = {};
    v8f Cr = wmma16(arm, B, z);
    v8f Ci = wmma16(aim, B, z);
    float m0=0.f, m1=0.f, m2=0.f, m3=0.f;
    for (int v = 0; v < 8; ++v){
      float pv = Cr[v]*Cr[v] + Ci[v]*Ci[v];     // |amp|^2, state M = v + 8*hi
      m0 += pv;
      m1 += ((v >> 2) & 1) ? -pv : pv;
      m2 += ((v >> 1) & 1) ? -pv : pv;
      m3 += (v & 1)        ? -pv : pv;
    }
    if (hi) m0 = -m0;                            // wire 0 sign = bit3(M) = lane half
    m0 += __shfl_xor(m0, 16, 32);
    m1 += __shfl_xor(m1, 16, 32);
    m2 += __shfl_xor(m2, 16, 32);
    m3 += __shfl_xor(m3, 16, 32);
    meas[pi][0]=m0; meas[pi][1]=m1; meas[pi][2]=m2; meas[pi][3]=m3;
  }

  // q_in[w*4 + p] = sigmoid(meas) * pi/2
  float qin[16];
  for (int p = 0; p < 4; ++p)
    for (int w = 0; w < 4; ++w)
      qin[w*4 + p] = 1.5707963267948966f / (1.f + __expf(-meas[p][w]));

  // ---- QFC encoder: per-qubit RY(a4).RX(a3).RZ(a2).RY(a1)|0>, tensor product ----
  C2 vq[4][2];
  for (int w = 0; w < 4; ++w){
    float a1 = qin[w], a2 = qin[4+w], a3 = qin[8+w], a4 = qin[12+w];
    float c1 = __cosf(a1*0.5f), s1 = __sinf(a1*0.5f);
    float cz = __cosf(a2*0.5f), sz = __sinf(a2*0.5f);
    C2 v0 = C2{c1*cz, -c1*sz};
    C2 v1 = C2{s1*cz,  s1*sz};
    float cx = __cosf(a3*0.5f), sx = __sinf(a3*0.5f);
    C2 w0 = C2{cx*v0.x + sx*v1.y,  cx*v0.y - sx*v1.x};
    C2 w1 = C2{sx*v0.y + cx*v1.x, -sx*v0.x + cx*v1.y};
    float cy = __cosf(a4*0.5f), sy = __sinf(a4*0.5f);
    vq[w][0] = C2{cy*w0.x - sy*w1.x, cy*w0.y - sy*w1.y};
    vq[w][1] = C2{sy*w0.x + cy*w1.x, sy*w0.y + cy*w1.y};
  }
  C2 t01c[4], t23c[4];
  for (int a = 0; a < 2; ++a)
    for (int b = 0; b < 2; ++b){
      t01c[a*2+b] = cmul(vq[0][a], vq[1][b]);
      t23c[a*2+b] = cmul(vq[2][a], vq[3][b]);
    }
  v16h Bq;
  for (int j = 0; j < 16; ++j){
    C2 ps = cmul(t01c[j >> 2], t23c[j & 3]);
    Bq[j] = (_Float16)(hi ? ps.y : ps.x);       // low lanes: Re, high lanes: Im (K=16..31)
  }
  v8f z = {};
  v8f Cr = wmma16(arq, Bq, z);
  v8f Ci = wmma16(aiq, Bq, z);
  float q0=0.f, q1=0.f, q2=0.f, q3=0.f;
  for (int v = 0; v < 8; ++v){
    float pv = Cr[v]*Cr[v] + Ci[v]*Ci[v];
    q0 += pv;
    q1 += ((v >> 2) & 1) ? -pv : pv;
    q2 += ((v >> 1) & 1) ? -pv : pv;
    q3 += (v & 1)        ? -pv : pv;
  }
  if (hi) q0 = -q0;
  q0 += __shfl_xor(q0, 16, 32);
  q1 += __shfl_xor(q1, 16, 32);
  q2 += __shfl_xor(q2, 16, 32);
  q3 += __shfl_xor(q3, 16, 32);
  float qo[4] = {q0, q1, q2, q3};

  // ---- final linear (4 -> 3) + log_softmax ----
  float lg[3];
  for (int k = 0; k < 3; ++k){
    float s = bv[k];
    for (int w = 0; w < 4; ++w) s += qo[w] * Wm[k*4 + w];
    lg[k] = s;
  }
  float mx = fmaxf(lg[0], fmaxf(lg[1], lg[2]));
  float se = __expf(lg[0]-mx) + __expf(lg[1]-mx) + __expf(lg[2]-mx);
  float lse = mx + __logf(se);
  if (!hi && valid){
    size_t o = (size_t)e * 3;
    out[o+0] = lg[0] - lse;
    out[o+1] = lg[1] - lse;
    out[o+2] = lg[2] - lse;
  }
}

extern "C" void kernel_launch(void* const* d_in, const int* in_sizes, int n_in,
                              void* d_out, int out_size, void* d_ws, size_t ws_size,
                              hipStream_t stream) {
  (void)n_in; (void)out_size; (void)d_ws; (void)ws_size;
  const float* x       = (const float*)d_in[0];
  const float* mps_u3  = (const float*)d_in[1];
  const float* mps_cu3 = (const float*)d_in[2];
  const float* qfc_u3  = (const float*)d_in[3];
  const float* qfc_cu3 = (const float*)d_in[4];
  const float* Wm      = (const float*)d_in[5];
  const float* bv      = (const float*)d_in[6];
  float* out = (float*)d_out;

  int bsz = in_sizes[0] / 784;
  if (bsz <= 0) return;
  long tiles = ((long)bsz + 15) / 16;   // 16 batch elements per wave
  int wavesPerBlock = 8;                // 256 threads
  int blocks = (int)((tiles + wavesPerBlock - 1) / wavesPerBlock);
  qmnist_kernel<<<blocks, 256, 0, stream>>>(x, mps_u3, mps_cu3, qfc_u3, qfc_cu3,
                                            Wm, bv, out, bsz);
}
